// GPT2Attention_78967268704327
// MI455X (gfx1250) — matrix operand order repaired and checked by reading, not compiled, not run on hardware
//
#include <hip/hip_runtime.h>

// ---------------------------------------------------------------------------
// GPT-2 decode attention step for MI455X (gfx1250, wave32).
//   B=8, T=8192, D=1024, H=16, HD=64
// HBM-bound: ~256MB avg KV-cache streaming @ 23.3TB/s -> ~11us floor.
// GEMMs: v_wmma_f32_16x16x32_f16 (f32 accumulate), 16-wide N tiles per wave.
// Attention: fp32 flash-decode split-T, coalesced float2 streams + prefetch.
// ---------------------------------------------------------------------------

#define KB 8            // batch
#define KT 8192         // cache length
#define KD 1024         // model dim
#define KH 16           // heads
#define KHD 64          // head dim
#define NSPLIT 32       // T splits for flash-decode
#define CHUNK (KT / NSPLIT)
#define ROW2 ((KH * KHD) / 2)   // float2 stride between consecutive t rows

typedef __attribute__((ext_vector_type(16))) _Float16 v16h;
typedef __attribute__((ext_vector_type(8)))  float    v8f;

union HV { v16h v; _Float16 h[16]; };
union FV { v8f  v; float    f[8];  };

// One wave computes a 16(M, padded from 8) x 16(N) f32 tile of A[16xK] * B[KxN]
// via v_wmma_f32_16x16x32_f16, K stepped by 32.
// A-pad trick: rows 8-15 re-read rows 0-7 (finite garbage, discarded at store)
// so all loads are unconditional & vectorizable -> no exec-mask branching.
__device__ __forceinline__ v8f wmma_tile_gemm(const float* __restrict__ A, int lda,
                                              const float* __restrict__ Bm, int ldb,
                                              int n0, int K) {
    const int lane = threadIdx.x & 31;
    const int half = lane >> 4;     // 0: lanes 0-15, 1: lanes 16-31
    const int lm   = lane & 15;
    const float* __restrict__ arow = A + (size_t)(lm & 7) * lda;
    const float* __restrict__ bcol = Bm + n0 + lm;
    v8f acc = {0.f, 0.f, 0.f, 0.f, 0.f, 0.f, 0.f, 0.f};
    for (int kk = 0; kk < K; kk += 32) {
        HV ua, ub;
        // A 16x32 f16 fragment (ISA 7.12.2): lane's halves j=0..7 hold
        // k = kk+8*half+j, j=8..15 hold k = kk+16+8*half+(j-8): two 32B runs.
        const float4 a0 = ((const float4*)(arow + kk + 8 * half))[0];
        const float4 a1 = ((const float4*)(arow + kk + 8 * half))[1];
        const float4 a2 = ((const float4*)(arow + kk + 16 + 8 * half))[0];
        const float4 a3 = ((const float4*)(arow + kk + 16 + 8 * half))[1];
        ua.h[0]  = (_Float16)a0.x; ua.h[1]  = (_Float16)a0.y;
        ua.h[2]  = (_Float16)a0.z; ua.h[3]  = (_Float16)a0.w;
        ua.h[4]  = (_Float16)a1.x; ua.h[5]  = (_Float16)a1.y;
        ua.h[6]  = (_Float16)a1.z; ua.h[7]  = (_Float16)a1.w;
        ua.h[8]  = (_Float16)a2.x; ua.h[9]  = (_Float16)a2.y;
        ua.h[10] = (_Float16)a2.z; ua.h[11] = (_Float16)a2.w;
        ua.h[12] = (_Float16)a3.x; ua.h[13] = (_Float16)a3.y;
        ua.h[14] = (_Float16)a3.z; ua.h[15] = (_Float16)a3.w;
        // B 32x16 f16 fragment: col = lm, k = kk + 8*half + j (j<8), kk + 16 + 8*half + (j-8) (j>=8).
        // Coalesced across lanes: each j is 2 x 64B row segments.
#pragma unroll
        for (int j = 0; j < 16; ++j) {
            ub.h[j] = (_Float16)bcol[(size_t)(kk + j + 8 * half + (j >= 8 ? 8 : 0)) * ldb];
        }
        acc = __builtin_amdgcn_wmma_f32_16x16x32_f16(
            /*neg_a=*/false, ua.v, /*neg_b=*/false, ub.v,
            /*c_mod=*/(short)0, acc, /*reuse_a=*/false, /*reuse_b=*/false);
    }
    return acc;
}

// qkv = x @ W_attn + b_attn ; also scatter k_step/v_step columns to d_out.
// grid = 48 blocks x 128 threads (4 waves); 192 N-tiles of 16.
__global__ void qkv_kernel(const float* __restrict__ x, const float* __restrict__ W,
                           const float* __restrict__ bias, float* __restrict__ qkv,
                           float* __restrict__ out_k, float* __restrict__ out_v) {
    const int tile = blockIdx.x * (blockDim.x >> 5) + (threadIdx.x >> 5);
    const int n0 = tile * 16;
    FV u; u.v = wmma_tile_gemm(x, KD, W, 3 * KD, n0, KD);
    const int lane = threadIdx.x & 31;
    const int half = lane >> 4;
    const int n = n0 + (lane & 15);
    const float bv = bias[n];
#pragma unroll
    for (int r = 0; r < 8; ++r) {
        const int mrow = r + 8 * half;               // C/D layout (ISA 7.12.2)
        if (mrow < KB) {
            const float val = u.f[r] + bv;
            qkv[mrow * 3 * KD + n] = val;
            if (n >= 2 * KD)      out_v[mrow * KD + (n - 2 * KD)] = val;
            else if (n >= KD)     out_k[mrow * KD + (n - KD)] = val;
        }
    }
}

__device__ __forceinline__ void online_update(float s, float2 vv2,
                                              float& m, float& l, float& a0, float& a1) {
    const float mn   = fmaxf(m, s);
    const float corr = __expf(m - mn);
    const float p    = __expf(s - mn);
    l  = l  * corr + p;
    a0 = a0 * corr + p * vv2.x;
    a1 = a1 * corr + p * vv2.y;
    m = mn;
}

// Flash-decode partials: one wave per (b,h,split). lane owns dims {2*lane, 2*lane+1}.
// One b64 load per tensor per row (coalesced 256B/row), prefetch 16 rows ahead.
__global__ void attn_partial_kernel(const float* __restrict__ qkv,
                                    const float* __restrict__ past_k,
                                    const float* __restrict__ past_v,
                                    const int* __restrict__ idx,
                                    float* __restrict__ ml, float* __restrict__ accs) {
    const int pair  = blockIdx.x;                 // 0..127  (b*16+h)
    const int b     = pair >> 4, h = pair & 15;
    const int lane  = threadIdx.x & 31;
    const int split = blockIdx.y * (blockDim.x >> 5) + (threadIdx.x >> 5); // 0..31
    const int idxb  = idx[b];

    const int t0 = split * CHUNK;
    int tEnd = t0 + CHUNK;
    if (tEnd > idxb) tEnd = idxb;                 // row idxb handled after the loop

    const float scale = 0.125f;                   // 1/sqrt(64)
    float2 q2 = ((const float2*)(qkv + b * 3 * KD + h * KHD))[lane];
    q2.x *= scale; q2.y *= scale;

    const size_t base = (((size_t)b * KT + t0) * KH + h) * KHD;
    const float2* __restrict__ kp = (const float2*)(past_k + base) + lane;
    const float2* __restrict__ vp = (const float2*)(past_v + base) + lane;

    float m = -3.3895314e38f, l = 0.f, a0 = 0.f, a1 = 0.f;
#pragma unroll 2
    for (int t = t0; t < tEnd; ++t) {
        __builtin_prefetch((const void*)(kp + 16 * ROW2), 0, 0);  // global_prefetch_b8
        __builtin_prefetch((const void*)(vp + 16 * ROW2), 0, 0);
        const float2 kk2 = *kp;
        const float2 vv2 = *vp;
        kp += ROW2; vp += ROW2;
        float s = q2.x * kk2.x + q2.y * kk2.y;
#pragma unroll
        for (int o = 16; o > 0; o >>= 1) s += __shfl_xor(s, o, 32);
        online_update(s, vv2, m, l, a0, a1);
    }
    // Current-step row (cache "update" = row substitution), once per (b,h).
    if (idxb >= t0 && idxb < t0 + CHUNK) {
        const float2 kk2 = ((const float2*)(qkv + b * 3 * KD + KD + h * KHD))[lane];
        const float2 vv2 = ((const float2*)(qkv + b * 3 * KD + 2 * KD + h * KHD))[lane];
        float s = q2.x * kk2.x + q2.y * kk2.y;
#pragma unroll
        for (int o = 16; o > 0; o >>= 1) s += __shfl_xor(s, o, 32);
        online_update(s, vv2, m, l, a0, a1);
    }
    const int ps = pair * NSPLIT + split;
    ((float2*)(accs + (size_t)ps * KHD))[lane] = make_float2(a0, a1);
    if (lane == 0) { ml[2 * ps] = m; ml[2 * ps + 1] = l; }
}

// Combine splits: grid 128 blocks x 64 threads (thread = head dim).
__global__ void attn_reduce_kernel(const float* __restrict__ ml,
                                   const float* __restrict__ accs,
                                   float* __restrict__ o) {
    const int pair = blockIdx.x;
    const int d = threadIdx.x;
    const int b = pair >> 4, h = pair & 15;
    float mg = -3.3895314e38f;
#pragma unroll 4
    for (int s = 0; s < NSPLIT; ++s) mg = fmaxf(mg, ml[2 * (pair * NSPLIT + s)]);
    float L = 0.f, acc = 0.f;
#pragma unroll 4
    for (int s = 0; s < NSPLIT; ++s) {
        const int ps = pair * NSPLIT + s;
        const float corr = __expf(ml[2 * ps] - mg);
        L   += corr * ml[2 * ps + 1];
        acc += corr * accs[(size_t)ps * KHD + d];
    }
    o[b * KD + h * KHD + d] = acc / L;
}

// y = o @ W_proj + b_proj : grid 16 x 128 (64 wave tiles).
__global__ void proj_kernel(const float* __restrict__ o, const float* __restrict__ W,
                            const float* __restrict__ bias, float* __restrict__ y) {
    const int tile = blockIdx.x * (blockDim.x >> 5) + (threadIdx.x >> 5);
    const int n0 = tile * 16;
    FV u; u.v = wmma_tile_gemm(o, KD, W, KD, n0, KD);
    const int lane = threadIdx.x & 31;
    const int half = lane >> 4;
    const int n = n0 + (lane & 15);
    const float bv = bias[n];
#pragma unroll
    for (int r = 0; r < 8; ++r) {
        const int mrow = r + 8 * half;
        if (mrow < KB) y[mrow * KD + n] = u.f[r] + bv;
    }
}

extern "C" void kernel_launch(void* const* d_in, const int* in_sizes, int n_in,
                              void* d_out, int out_size, void* d_ws, size_t ws_size,
                              hipStream_t stream) {
    const float* x      = (const float*)d_in[0];
    const float* past_k = (const float*)d_in[1];
    const float* past_v = (const float*)d_in[2];
    const int*   idx    = (const int*)d_in[3];
    const float* W_attn = (const float*)d_in[4];
    const float* b_attn = (const float*)d_in[5];
    const float* W_proj = (const float*)d_in[6];
    const float* b_proj = (const float*)d_in[7];

    float* out = (float*)d_out;     // [y(8192) | k_step(8192) | v_step(8192)]
    float* ws  = (float*)d_ws;
    // Workspace layout (floats): total ~303K floats (~1.2 MB)
    float* qkv  = ws;                       // 8*3072       = 24576
    float* o    = ws + 24576;               // 8*1024       = 8192
    float* ml   = ws + 32768;               // 128*32*2     = 8192
    float* accs = ws + 40960;               // 128*32*64    = 262144

    qkv_kernel<<<48, 128, 0, stream>>>(x, W_attn, b_attn, qkv, out + 8192, out + 16384);
    attn_partial_kernel<<<dim3(128, 4), 256, 0, stream>>>(qkv, past_k, past_v, idx, ml, accs);
    attn_reduce_kernel<<<128, 64, 0, stream>>>(ml, accs, o);
    proj_kernel<<<16, 128, 0, stream>>>(o, W_proj, b_proj, out);
}
